// TestModel_30116310680263
// MI455X (gfx1250) — matrix operand order repaired and checked by reading, not compiled, not run on hardware
//
#include <hip/hip_runtime.h>

// CDNA5 / gfx1250, wave32. WMMA bf16 path: v_wmma_f32_16x16x32_bf16.
typedef __attribute__((ext_vector_type(16))) __bf16 v16bf;
typedef __attribute__((ext_vector_type(8)))  float  v8f;

#define B_ 16
#define T_ 4096
#define D_ 512
#define K_ 1024   // 2*D
#define L_ 12
#define WFRAG_PER_LAYER (32 * 32 * 32 * 8)   // kb(32) x ntile(32) x lane(32) x 8 dwords = 2^18
#define N_LAYERS 25                          // 12 up + 12 down + 1 leaf

// f32 -> bf16 (round-to-nearest-even), packed pair into one dword.
__device__ __forceinline__ unsigned int pack2bf(float a, float b) {
  unsigned int ua = __float_as_uint(a);
  unsigned int ub = __float_as_uint(b);
  ua = ua + 0x7fffu + ((ua >> 16) & 1u);
  ub = ub + 0x7fffu + ((ub >> 16) & 1u);
  return (ua >> 16) | (ub & 0xffff0000u);
}

// One-shot: convert all 25 weight matrices (512x1024 f32, row-major K-contiguous)
// into bf16 stored in WMMA B-fragment layout:
//   dword idx = ((layer*32 + kb)*32 + nt)*32*8 + lane*8 + d
//   n = nt*16 + (lane&15),  k = kb*32 + (lane>>4)*16 + 2*d  (e = k&15, half = k>=16)
__global__ __launch_bounds__(256) void wconv_kernel(
    const float* __restrict__ up_W, const float* __restrict__ down_W,
    const float* __restrict__ leaf_W, unsigned int* __restrict__ wfrag,
    long long total)
{
  long long idx = (long long)blockIdx.x * blockDim.x + threadIdx.x;
  if (idx >= total) return;
  int d     = (int)(idx & 7);
  int lane  = (int)((idx >> 3) & 31);
  int nt    = (int)((idx >> 8) & 31);
  int kb    = (int)((idx >> 13) & 31);
  int layer = (int)(idx >> 18);
  int n = nt * 16 + (lane & 15);
  int k = kb * 32 + ((lane >> 4) << 4) + (d << 1);
  const float* Wsrc;
  if (layer < 12)      Wsrc = up_W   + (size_t)layer * D_ * K_;
  else if (layer < 24) Wsrc = down_W + (size_t)(layer - 12) * D_ * K_;
  else                 Wsrc = leaf_W;
  const float* s = Wsrc + (size_t)n * K_ + k;
  wfrag[idx] = pack2bf(s[0], s[1]);
}

// Generic  Y[r][n] = LIF( sum_k A[r][k] * W[n][k] + bias[n] , thr[n] )
// A[r][k] = (k < 512) ? p0[r*s0 + k] : p1[r*s1 + (k-512)]
// Block tile: M=64 rows x N=512 cols (full row -> LIF reduction block-local).
// 8 waves; wave w covers columns [w*64, w*64+64): 4 n-tiles x 4 m-tiles = 16 WMMA/K-step.
// B fragments come pre-converted (bf16, fragment layout) straight from global/L2.
__global__ __launch_bounds__(256) void gemm_lif_kernel(
    const float* __restrict__ p0, long long s0,
    const float* __restrict__ p1, long long s1,
    const unsigned int* __restrict__ Wf,   // this layer's fragment weights
    const float* __restrict__ bias,        // (512)
    const float* __restrict__ thr,         // (512)
    float* __restrict__ out, long long outStride,
    long long Mtot)
{
  __shared__ __align__(32) unsigned int AsU[4 * 32 * 8];   // 4 m-tiles, 4KB
  __shared__ float rowCnt[64];

  const int tid  = threadIdx.x;
  const int lane = tid & 31;
  const int wv   = tid >> 5;
  const long long rowBase = (long long)blockIdx.x * 64;

  v8f zero8 = {0.f, 0.f, 0.f, 0.f, 0.f, 0.f, 0.f, 0.f};
  v8f acc[4][4];
  #pragma unroll
  for (int i = 0; i < 4; ++i)
    #pragma unroll
    for (int j = 0; j < 4; ++j) acc[i][j] = zero8;

  for (int kb = 0; kb < 32; ++kb) {
    // ---- stage A tile (64 rows x 32 k), f32 -> bf16, fragment layout ----
    #pragma unroll
    for (int t = 0; t < 2; ++t) {
      int chunk = tid + t * 256;     // 0..511
      int m  = chunk >> 3;           // 0..63
      int kc = (chunk & 7) << 2;     // 0,4,...,28
      long long gr = rowBase + m;
      float4 f = {0.f, 0.f, 0.f, 0.f};
      if (gr < Mtot) {
        int kk = kb * 32 + kc;
        const float* src = (kk < 512) ? (p0 + gr * s0 + kk)
                                      : (p1 + gr * s1 + (kk - 512));
        f = *(const float4*)src;
      }
      // A-frag mapping (16-bit A 16x32): lane = half*16 + m%16, e = k&15, half = (k>=16)
      int half = (kc >> 4) & 1;
      int e    = kc & 15;
      int la   = half * 16 + (m & 15);
      int mt   = m >> 4;
      unsigned int* dst = &AsU[(mt * 32 + la) * 8 + (e >> 1)];
      dst[0] = pack2bf(f.x, f.y);
      dst[1] = pack2bf(f.z, f.w);
    }
    __syncthreads();

    v16bf a[4];
    #pragma unroll
    for (int mt = 0; mt < 4; ++mt)
      a[mt] = *(const v16bf*)&AsU[(mt * 32 + lane) * 8];

    #pragma unroll
    for (int nt = 0; nt < 4; ++nt) {
      const unsigned int* bp =
          Wf + ((size_t)(kb * 32 + (wv * 4 + nt)) * 32 + lane) * 8;
      v16bf b = *(const v16bf*)bp;   // 32B from L2 -> 2x global_load_b128
      #pragma unroll
      for (int mt = 0; mt < 4; ++mt)
        acc[mt][nt] = __builtin_amdgcn_wmma_f32_16x16x32_bf16(
            false, a[mt], false, b, (short)0, acc[mt][nt], false, false);
    }
    __syncthreads();
  }

  // ---- LIF epilogue: spike = (y >= thr); out = spike / (rowSum + 1e-8) ----
  if (tid < 64) rowCnt[tid] = 0.0f;
  __syncthreads();

  const int nl   = lane & 15;
  const int rOff = (lane >> 4) * 8;   // C/D layout: lanes 16-31 hold rows 8..15

  #pragma unroll
  for (int mt = 0; mt < 4; ++mt)
    #pragma unroll
    for (int nt = 0; nt < 4; ++nt) {
      int n = wv * 64 + nt * 16 + nl;
      float bv = bias[n];
      float tv = thr[n];
      #pragma unroll
      for (int r = 0; r < 8; ++r) {
        float y = acc[mt][nt][r] + bv;
        acc[mt][nt][r] = (y >= tv) ? 1.0f : 0.0f;   // reuse accs as spikes
      }
    }
  #pragma unroll
  for (int mt = 0; mt < 4; ++mt)
    #pragma unroll
    for (int r = 0; r < 8; ++r) {
      float c = acc[mt][0][r] + acc[mt][1][r] + acc[mt][2][r] + acc[mt][3][r];
      atomicAdd(&rowCnt[mt * 16 + rOff + r], c);    // ds_add_f32
    }
  __syncthreads();

  #pragma unroll
  for (int mt = 0; mt < 4; ++mt)
    #pragma unroll
    for (int r = 0; r < 8; ++r) {
      long long gr = rowBase + mt * 16 + rOff + r;
      if (gr < Mtot) {
        float inv = 1.0f / (rowCnt[mt * 16 + rOff + r] + 1e-8f);
        #pragma unroll
        for (int nt = 0; nt < 4; ++nt) {
          int n = wv * 64 + nt * 16 + nl;
          out[gr * outStride + n] = acc[mt][nt][r] * inv;
        }
      }
    }
}

// ctx_new[2r] = ctx_old[r]  (down-pass interleave, even slots), float4 granularity
__global__ void copy_even_kernel(const float* __restrict__ src,
                                 float* __restrict__ dst, long long n4)
{
  long long i = (long long)blockIdx.x * blockDim.x + threadIdx.x;
  if (i >= n4) return;
  long long row = i >> 7;          // D/4 = 128 float4 per row
  long long c   = i & 127;
  ((float4*)dst)[row * 256 + c] = ((const float4*)src)[row * 128 + c];
}

__global__ void zero_kernel(float* __restrict__ p, long long n)
{
  long long i = (long long)blockIdx.x * blockDim.x + threadIdx.x;
  if (i < n) p[i] = 0.0f;
}

extern "C" void kernel_launch(void* const* d_in, const int* in_sizes, int n_in,
                              void* d_out, int out_size, void* d_ws, size_t ws_size,
                              hipStream_t stream) {
  (void)in_sizes; (void)n_in; (void)out_size; (void)ws_size;
  const float* x        = (const float*)d_in[0];
  const float* up_W     = (const float*)d_in[1];
  const float* up_b     = (const float*)d_in[2];
  const float* up_thr   = (const float*)d_in[3];
  const float* down_W   = (const float*)d_in[4];
  const float* down_b   = (const float*)d_in[5];
  const float* down_thr = (const float*)d_in[6];
  const float* leaf_W   = (const float*)d_in[7];
  const float* leaf_b   = (const float*)d_in[8];
  const float* leaf_thr = (const float*)d_in[9];
  float* out = (float*)d_out;
  float* ws  = (float*)d_ws;

  // Workspace layout (floats): summaries[1..12], ctxA (B*T*D), ctxB (B*T/2*D),
  // then bf16 weight fragments (25 * 2^18 dwords).
  float* summ[L_ + 1];
  size_t off = 0;
  summ[0] = nullptr; // level 0 is x itself
  for (int l = 1; l <= L_; ++l) {
    summ[l] = ws + off;
    off += (size_t)B_ * (T_ >> l) * D_;
  }
  float* ctxA = ws + off; off += (size_t)B_ * T_ * D_;
  float* ctxB = ws + off; off += (size_t)B_ * (T_ / 2) * D_;
  unsigned int* wfrag = (unsigned int*)(ws + off);

  // ---------------- Weight pre-conversion (f32 -> bf16 fragment layout) ----
  {
    long long total = (long long)N_LAYERS * WFRAG_PER_LAYER;
    int blocks = (int)((total + 255) / 256);
    wconv_kernel<<<blocks, 256, 0, stream>>>(up_W, down_W, leaf_W, wfrag, total);
  }
  const unsigned int* upWf   = wfrag;
  const unsigned int* downWf = wfrag + (size_t)12 * WFRAG_PER_LAYER;
  const unsigned int* leafWf = wfrag + (size_t)24 * WFRAG_PER_LAYER;

  // ---------------- Up pass ----------------
  const float* cur = x;
  int Tc = T_;
  for (int l = 0; l < L_; ++l) {
    int P = Tc / 2;
    long long M = (long long)B_ * P;
    int blocks = (int)((M + 63) / 64);
    gemm_lif_kernel<<<blocks, 256, 0, stream>>>(
        cur, 2LL * D_,            // k<512:  cur[2r]
        cur + D_, 2LL * D_,       // k>=512: cur[2r+1]
        upWf + (size_t)l * WFRAG_PER_LAYER,
        up_b + (size_t)l * D_, up_thr + (size_t)l * D_,
        summ[l + 1], (long long)D_, M);
    cur = summ[l + 1];
    Tc = P;
  }

  // ---------------- Down pass ----------------
  zero_kernel<<<(B_ * D_ + 255) / 256, 256, 0, stream>>>(ctxA, (long long)B_ * D_);
  float* ctx_old = ctxA;
  long long Pc = 1;
  for (int i = L_ - 1; i >= 0; --i) {
    float* ctx_new = (ctx_old == ctxA) ? ctxB : ctxA;
    long long M = (long long)B_ * Pc;
    const float* psum = (i == 0) ? x : summ[i];
    int blocks = (int)((M + 63) / 64);
    // right -> odd rows of ctx_new
    gemm_lif_kernel<<<blocks, 256, 0, stream>>>(
        ctx_old, (long long)D_,   // k<512:  ctx[r]
        psum, 2LL * D_,           // k>=512: summaries[i][2r] (left child)
        downWf + (size_t)i * WFRAG_PER_LAYER,
        down_b + (size_t)i * D_, down_thr + (size_t)i * D_,
        ctx_new + D_, 2LL * D_, M);
    // old ctx -> even rows of ctx_new
    long long n4 = M * D_ / 4;
    copy_even_kernel<<<(int)((n4 + 255) / 256), 256, 0, stream>>>(ctx_old, ctx_new, n4);
    ctx_old = ctx_new;
    Pc *= 2;
  }
  // ctx_old now holds (B, T, D) context (12 steps from ctxA end in ctxA, the big buffer)

  // ---------------- Leaf ----------------
  {
    long long M = (long long)B_ * T_;
    int blocks = (int)((M + 63) / 64);
    gemm_lif_kernel<<<blocks, 256, 0, stream>>>(
        x, (long long)D_,         // k<512:  x[r]
        ctx_old, (long long)D_,   // k>=512: ctx[r]
        leafWf, leaf_b, leaf_thr,
        out, (long long)D_, M);
  }
}
